// M3GNet_89532888252971
// MI455X (gfx1250) — compile-verified, matrix operand order checked by reading, not run on hardware
//
#include <hip/hip_runtime.h>
#include <hip/hip_bf16.h>

// ---------------------------------------------------------------------------
// M3GNet forward for MI455X (gfx1250, wave32, WMMA).
// All H=64 matmuls run on v_wmma_f32_16x16x32_f16 (f16 in, f32 accumulate).
// Weights pre-packed into WMMA-B fragment layout (1 v16h load per lane/frag).
// Activations kept f32 in HBM; per-wave LDS staging for gMLP features.
// Segment sums via global f32 atomics.
// ---------------------------------------------------------------------------

typedef __attribute__((ext_vector_type(16))) _Float16 v16h;
typedef __attribute__((ext_vector_type(8)))  float    v8f;

#define H 64
#define NLAYERS 4
#define WPB 4   // waves (tiles) per block -> 128 threads

// ---- WMMA fragment helpers -------------------------------------------------

// half index h (0..15) + lane -> K offset within a 32-wide K chunk
// (per CDNA5 ISA 16-bit A layout: lanes 0-15 hold K 0-7/16-23, lanes 16-31 hold
// K 8-15/24-31; we use the same mapping for B with N = lane&15)
__device__ __forceinline__ int frag_k(int lane, int h) {
  return ((h >> 3) << 4) | ((lane >> 4) << 3) | (h & 7);
}

__device__ __forceinline__ v8f zero8() {
  v8f z;
#pragma unroll
  for (int i = 0; i < 8; ++i) z[i] = 0.0f;
  return z;
}

__device__ __forceinline__ v8f wmma16(v16h a, v16h b, v8f c) {
  // D = A(16x32 f16) * B(32x16 f16) + C(16x16 f32)
  return __builtin_amdgcn_wmma_f32_16x16x32_f16(false, a, false, b, (short)0, c,
                                                false, false);
}

// B fragment from pre-packed buffer: one coalesced 32B load per lane
__device__ __forceinline__ v16h frag_b_pack(const _Float16* pack, int frag, int lane) {
  return *(const v16h*)(pack + ((size_t)frag * 32 + lane) * 16);
}

// A fragment gathered from a row-major f32 matrix (per-lane row index)
__device__ __forceinline__ v16h frag_a_g32(const float* mat, int row, int ld,
                                           int k0, int lane) {
  const float* p = mat + (size_t)row * ld + k0;
  v16h a;
#pragma unroll
  for (int h = 0; h < 16; ++h) a[h] = (_Float16)p[frag_k(lane, h)];
  return a;
}

// A fragment from an f16 LDS staging buffer
__device__ __forceinline__ v16h frag_a_lds(const _Float16* buf, int row, int ld,
                                           int k0, int lane) {
  const _Float16* p = buf + row * ld + k0;
  v16h a;
#pragma unroll
  for (int h = 0; h < 16; ++h) a[h] = p[frag_k(lane, h)];
  return a;
}

__device__ __forceinline__ float sigm(float x) { return 1.0f / (1.0f + __expf(-x)); }
__device__ __forceinline__ float siluf(float x) { return x * sigm(x); }

// ---- weight pre-pack: f32 [K,64] -> WMMA-B fragment order f16 ---------------
__global__ void prepack_kernel(const float* __restrict__ W, _Float16* __restrict__ out,
                               int K, int Kpad) {
  int idx = blockIdx.x * blockDim.x + threadIdx.x;
  int total = (Kpad / 32) * 4 * 32;
  if (idx >= total) return;
  int lane = idx & 31;
  int f = idx >> 5;
  int nt = f & 3;
  int kc = f >> 2;
  int n = nt * 16 + (lane & 15);
  _Float16* o = out + (size_t)idx * 16;
#pragma unroll
  for (int h = 0; h < 16; ++h) {
    int k = kc * 32 + frag_k(lane, h);
    o[h] = (k < K) ? (_Float16)W[(size_t)k * H + n] : (_Float16)0.0f;
  }
}

// ---- edge geometry + smooth radial Bessel + cutoff --------------------------
__global__ void edge_geom_kernel(const float* __restrict__ pos,
                                 const float* __restrict__ cell,
                                 const float* __restrict__ pbc,
                                 const int* __restrict__ ei0,
                                 const int* __restrict__ ei1,
                                 float* __restrict__ evec, float* __restrict__ elen,
                                 float* __restrict__ e0b, float* __restrict__ fcr,
                                 int E, int APG) {
  int e = blockIdx.x * blockDim.x + threadIdx.x;
  if (e >= E) return;
  int i0 = ei0[e], i1 = ei1[e];
  int g = i0 / APG;
  const float* C = cell + (size_t)g * 9;
  float p0 = pbc[(size_t)e * 3], p1 = pbc[(size_t)e * 3 + 1], p2 = pbc[(size_t)e * 3 + 2];
  float ox = p0 * C[0] + p1 * C[3] + p2 * C[6];
  float oy = p0 * C[1] + p1 * C[4] + p2 * C[7];
  float oz = p0 * C[2] + p1 * C[5] + p2 * C[8];
  float vx = pos[(size_t)i0 * 3 + 0] - (pos[(size_t)i1 * 3 + 0] + ox);
  float vy = pos[(size_t)i0 * 3 + 1] - (pos[(size_t)i1 * 3 + 1] + oy);
  float vz = pos[(size_t)i0 * 3 + 2] - (pos[(size_t)i1 * 3 + 2] + oz);
  float r = sqrtf(vx * vx + vy * vy + vz * vz);
  evec[(size_t)e * 3 + 0] = vx;
  evec[(size_t)e * 3 + 1] = vy;
  evec[(size_t)e * 3 + 2] = vz;
  elen[e] = r;

  const float PI_ = 3.14159265358979f;
  float fs[4], en[4];
#pragma unroll
  for (int n = 0; n < 4; ++n) {
    float n1 = n + 1.0f, n2 = n + 2.0f;
    float c = ((n & 1) ? -1.0f : 1.0f) * 1.41421356f * PI_ / 11.18033989f *
              (n1 * n2 / sqrtf(n1 * n1 + n2 * n2));
    float x1 = r * n1 * PI_ / 5.0f, x2 = r * n2 * PI_ / 5.0f;
    fs[n] = c * (sinf(x1) / x1 + sinf(x2) / x2);
    float n1_4 = n1 * n1 * n1 * n1;
    en[n] = (float)(n * n) * n2 * n2 / (4.0f * n1_4 + 1.0f);
  }
  float gprev = fs[0], dprev = 1.0f;
  e0b[(size_t)e * 4 + 0] = gprev;
#pragma unroll
  for (int n = 1; n < 4; ++n) {
    float d = 1.0f - en[n] / dprev;
    float gn = (fs[n] + sqrtf(en[n] / dprev) * gprev) / sqrtf(d);
    e0b[(size_t)e * 4 + n] = gn;
    gprev = gn;
    dprev = d;
  }
  float u = r * 0.25f;
  float u2 = u * u, u3 = u2 * u;
  float f = 1.0f - 6.0f * u3 * u2 + 15.0f * u2 * u2 - 10.0f * u3;
  fcr[e] = (r < 4.0f) ? f : 0.0f;
}

// e = e0 @ W_edge_enc (tiny K=4 -> scalar dot per output element)
__global__ void edge_enc_kernel(const float* __restrict__ e0b,
                                const float* __restrict__ Wenc,
                                float* __restrict__ efeat, int total) {
  int idx = blockIdx.x * blockDim.x + threadIdx.x;
  if (idx >= total) return;
  int e = idx >> 6, h = idx & 63;
  float s = 0.0f;
#pragma unroll
  for (int n = 0; n < 4; ++n) s += e0b[(size_t)e * 4 + n] * Wenc[n * H + h];
  efeat[idx] = s;
}

// v = W_embed[z]
__global__ void atom_init_kernel(const float* __restrict__ Wemb,
                                 const int* __restrict__ attr,
                                 float* __restrict__ v, int total) {
  int idx = blockIdx.x * blockDim.x + threadIdx.x;
  if (idx >= total) return;
  int a = idx >> 6, h = idx & 63;
  v[idx] = Wemb[(size_t)attr[a] * H + h];
}

// ---- triple precompute: angles, spherical basis, cutoff product -------------
__device__ __constant__ float SBZ[16] = {
    3.14159265f, 6.28318531f, 9.42477796f,  12.56637061f,
    4.49340946f, 7.72525184f, 10.90412166f, 14.06619391f,
    5.76345920f, 9.09501133f, 12.32294097f, 15.51460301f,
    6.98793200f, 10.41711855f, 13.69802315f, 16.92362129f};

__device__ __forceinline__ float sph_jl(float x, int l) {
  x = fmaxf(x, 1e-6f);
  float s = sinf(x), c = cosf(x);
  if (l == 0) return s / x;
  if (l == 1) return s / (x * x) - c / x;
  if (l == 2) return (3.0f / (x * x * x) - 1.0f / x) * s - 3.0f * c / (x * x);
  float x2 = x * x;
  return (15.0f / (x2 * x2) - 6.0f / x2) * s - (15.0f / (x2 * x) - 1.0f / x) * c;
}

__device__ __forceinline__ float legendre_(float c, int l) {
  if (l == 0) return 1.0f;
  if (l == 1) return c;
  if (l == 2) return 0.5f * (3.0f * c * c - 1.0f);
  return 0.5f * (5.0f * c * c * c - 3.0f * c);
}

__global__ void triple_init_kernel(const int* __restrict__ tbi,
                                   const float* __restrict__ evec,
                                   const float* __restrict__ elen,
                                   const float* __restrict__ fcr,
                                   const int* __restrict__ ei1,
                                   int* __restrict__ t_ij, int* __restrict__ t_ak,
                                   float* __restrict__ sbf, float* __restrict__ fc3,
                                   int T, int TPG, int BPG) {
  int t = blockIdx.x * blockDim.x + threadIdx.x;
  if (t >= T) return;
  int g = t / TPG;
  int bias = g * BPG;
  int ij = tbi[2 * t + 0] + bias;
  int ik = tbi[2 * t + 1] + bias;
  float ax = evec[(size_t)ij * 3 + 0], ay = evec[(size_t)ij * 3 + 1], az = evec[(size_t)ij * 3 + 2];
  float bx = evec[(size_t)ik * 3 + 0], by = evec[(size_t)ik * 3 + 1], bz = evec[(size_t)ik * 3 + 2];
  float rij = elen[ij], rik = elen[ik];
  float cth = (ax * bx + ay * by + az * bz) / (rij * rik);
  cth = fminf(fmaxf(cth, -1.0f + 1e-7f), 1.0f - 1e-7f);
#pragma unroll
  for (int l = 0; l < 4; ++l) {
    float pl = legendre_(cth, l);
#pragma unroll
    for (int n = 0; n < 4; ++n)
      sbf[(size_t)t * 16 + l * 4 + n] = sph_jl(SBZ[l * 4 + n] * rik * 0.2f, l) * pl;
  }
  fc3[t] = fcr[ij] * fcr[ik];
  t_ij[t] = ij;
  t_ak[t] = ei1[ik];
}

// ---- per-layer triple message: WMMA + scatter -------------------------------
// msg = (sbf3 @ W3_sbf) * sigmoid(v[ak] @ W3_atom) * fc3 ; agg[ij] += msg
__global__ __launch_bounds__(WPB * 32) void triple_kernel(
    const float* __restrict__ v_in, const float* __restrict__ sbf,
    const float* __restrict__ fc3, const int* __restrict__ t_ij,
    const int* __restrict__ t_ak, float* __restrict__ agg,
    const _Float16* __restrict__ pw3atom, const _Float16* __restrict__ pw3sbf,
    int T) {
  int wave = threadIdx.x >> 5, lane = threadIdx.x & 31;
  int t0 = (blockIdx.x * WPB + wave) * 16;
  if (t0 >= T) return;

  v8f accS[4], accP[4];
#pragma unroll
  for (int nt = 0; nt < 4; ++nt) { accS[nt] = zero8(); accP[nt] = zero8(); }

  int arow = t_ak[t0 + (lane & 15)];
#pragma unroll
  for (int kc = 0; kc < 2; ++kc) {
    v16h a = frag_a_g32(v_in, arow, H, kc * 32, lane);
#pragma unroll
    for (int nt = 0; nt < 4; ++nt)
      accS[nt] = wmma16(a, frag_b_pack(pw3atom, kc * 4 + nt, lane), accS[nt]);
  }
  {
    const float* p = sbf + (size_t)(t0 + (lane & 15)) * 16;
    v16h a;
#pragma unroll
    for (int h = 0; h < 16; ++h) {
      int k = frag_k(lane, h);
      a[h] = (k < 16) ? (_Float16)p[k] : (_Float16)0.0f;
    }
#pragma unroll
    for (int nt = 0; nt < 4; ++nt)
      accP[nt] = wmma16(a, frag_b_pack(pw3sbf, nt, lane), accP[nt]);
  }

  int rb = (lane >> 4) << 3, cb = lane & 15;
#pragma unroll
  for (int r = 0; r < 8; ++r) {
    int m = t0 + rb + r;
    float f = fc3[m];
    int ij = t_ij[m];
#pragma unroll
    for (int nt = 0; nt < 4; ++nt) {
      float msg = accP[nt][r] * sigm(accS[nt][r]) * f;
      atomicAdd(&agg[(size_t)ij * H + nt * 16 + cb], msg);
    }
  }
}

// ---- gated MLP on a 16-row tile, executed by one wave -----------------------
__device__ __forceinline__ void gmlp_wave(const _Float16 (*feat)[192],
                                          _Float16 (*hbuf)[64],
                                          const _Float16* p1, const _Float16* p2,
                                          v8f out[4], int lane) {
  v8f acc[4];
#pragma unroll
  for (int nt = 0; nt < 4; ++nt) acc[nt] = zero8();
#pragma unroll
  for (int kc = 0; kc < 6; ++kc) {
    v16h a = frag_a_lds(&feat[0][0], lane & 15, 192, kc * 32, lane);
#pragma unroll
    for (int nt = 0; nt < 4; ++nt)
      acc[nt] = wmma16(a, frag_b_pack(p1, kc * 4 + nt, lane), acc[nt]);
  }
  int rb = (lane >> 4) << 3, cb = lane & 15;
#pragma unroll
  for (int nt = 0; nt < 4; ++nt)
#pragma unroll
    for (int r = 0; r < 8; ++r)
      hbuf[rb + r][nt * 16 + cb] = (_Float16)siluf(acc[nt][r]);
#pragma unroll
  for (int nt = 0; nt < 4; ++nt) out[nt] = zero8();
#pragma unroll
  for (int kc = 0; kc < 2; ++kc) {
    v16h a = frag_a_lds(&hbuf[0][0], lane & 15, 64, kc * 32, lane);
#pragma unroll
    for (int nt = 0; nt < 4; ++nt)
      out[nt] = wmma16(a, frag_b_pack(p2, kc * 4 + nt, lane), out[nt]);
  }
}

struct EdgeW {
  const _Float16 *w3edge, *wb1, *wb2, *wbg1, *wbg2, *wa1, *wa2, *wag1, *wag2;
  const float *we0b, *we0a;
};

// ---- fused per-layer edge kernel -------------------------------------------
__global__ __launch_bounds__(WPB * 32) void edge_layer_kernel(
    const int* __restrict__ ei0, const int* __restrict__ ei1,
    const float* __restrict__ agg, float* __restrict__ efeat,
    const float* __restrict__ v_in, float* __restrict__ v_out,
    const float* __restrict__ e0b, EdgeW w, int E) {
  __shared__ _Float16 feat_s[WPB][16][192];
  __shared__ _Float16 h_s[WPB][16][64];
  __shared__ int i0_s[WPB][16], i1_s[WPB][16];

  int wave = threadIdx.x >> 5, lane = threadIdx.x & 31;
  int t0 = (blockIdx.x * WPB + wave) * 16;
  if (t0 >= E) return;

  if (lane < 16) {
    i0_s[wave][lane] = ei0[t0 + lane];
    i1_s[wave][lane] = ei1[t0 + lane];
  }
  int rb = (lane >> 4) << 3, cb = lane & 15;

  // ---- e += silu(agg @ W3_edge) ----
  v8f acc[4];
#pragma unroll
  for (int nt = 0; nt < 4; ++nt) acc[nt] = zero8();
#pragma unroll
  for (int kc = 0; kc < 2; ++kc) {
    v16h a = frag_a_g32(agg, t0 + (lane & 15), H, kc * 32, lane);
#pragma unroll
    for (int nt = 0; nt < 4; ++nt)
      acc[nt] = wmma16(a, frag_b_pack(w.w3edge, kc * 4 + nt, lane), acc[nt]);
  }
#pragma unroll
  for (int nt = 0; nt < 4; ++nt)
#pragma unroll
    for (int r = 0; r < 8; ++r) {
      int m = rb + r, col = nt * 16 + cb;
      size_t gi = (size_t)(t0 + m) * H + col;
      float e = efeat[gi] + siluf(acc[nt][r]);
      efeat[gi] = e;
      feat_s[wave][m][128 + col] = (_Float16)e;
    }

  // ---- stage gathered v[i0], v[i1] into feat ----
  for (int idx = lane; idx < 16 * H; idx += 32) {
    int r = idx >> 6, c = idx & 63;
    feat_s[wave][r][c] = (_Float16)v_in[(size_t)i0_s[wave][r] * H + c];
    feat_s[wave][r][64 + c] = (_Float16)v_in[(size_t)i1_s[wave][r] * H + c];
  }

  v8f am[4], ag[4];
  // ---- bond gMLP ----
  gmlp_wave(feat_s[wave], h_s[wave], w.wb1, w.wb2, am, lane);
  gmlp_wave(feat_s[wave], h_s[wave], w.wbg1, w.wbg2, ag, lane);
#pragma unroll
  for (int nt = 0; nt < 4; ++nt)
#pragma unroll
    for (int r = 0; r < 8; ++r) {
      int m = rb + r, col = nt * 16 + cb;
      float ew = 0.0f;
#pragma unroll
      for (int n = 0; n < 4; ++n) ew += e0b[(size_t)(t0 + m) * 4 + n] * w.we0b[n * H + col];
      size_t gi = (size_t)(t0 + m) * H + col;
      float e = efeat[gi] + am[nt][r] * sigm(ag[nt][r]) * ew;
      efeat[gi] = e;
      feat_s[wave][m][128 + col] = (_Float16)e;
    }

  // ---- atom gMLP + scatter to v ----
  gmlp_wave(feat_s[wave], h_s[wave], w.wa1, w.wa2, am, lane);
  gmlp_wave(feat_s[wave], h_s[wave], w.wag1, w.wag2, ag, lane);
#pragma unroll
  for (int nt = 0; nt < 4; ++nt)
#pragma unroll
    for (int r = 0; r < 8; ++r) {
      int m = rb + r, col = nt * 16 + cb;
      float ew = 0.0f;
#pragma unroll
      for (int n = 0; n < 4; ++n) ew += e0b[(size_t)(t0 + m) * 4 + n] * w.we0a[n * H + col];
      float mv = am[nt][r] * sigm(ag[nt][r]) * ew;
      atomicAdd(&v_out[(size_t)i0_s[wave][m] * H + col], mv);
    }
}

// ---- final readout: 2-layer silu MLP (WMMA) + H->1 dot + graph reduce ------
__device__ __forceinline__ void mlp2_wave(const float* v_in, int a0,
                                          const _Float16* p1, const _Float16* p2,
                                          _Float16 (*hbuf)[64], int lane) {
  v8f acc[4];
#pragma unroll
  for (int nt = 0; nt < 4; ++nt) acc[nt] = zero8();
#pragma unroll
  for (int kc = 0; kc < 2; ++kc) {
    v16h a = frag_a_g32(v_in, a0 + (lane & 15), H, kc * 32, lane);
#pragma unroll
    for (int nt = 0; nt < 4; ++nt)
      acc[nt] = wmma16(a, frag_b_pack(p1, kc * 4 + nt, lane), acc[nt]);
  }
  int rb = (lane >> 4) << 3, cb = lane & 15;
#pragma unroll
  for (int nt = 0; nt < 4; ++nt)
#pragma unroll
    for (int r = 0; r < 8; ++r)
      hbuf[rb + r][nt * 16 + cb] = (_Float16)siluf(acc[nt][r]);
  v8f acc2[4];
#pragma unroll
  for (int nt = 0; nt < 4; ++nt) acc2[nt] = zero8();
#pragma unroll
  for (int kc = 0; kc < 2; ++kc) {
    v16h a = frag_a_lds(&hbuf[0][0], lane & 15, 64, kc * 32, lane);
#pragma unroll
    for (int nt = 0; nt < 4; ++nt)
      acc2[nt] = wmma16(a, frag_b_pack(p2, kc * 4 + nt, lane), acc2[nt]);
  }
#pragma unroll
  for (int nt = 0; nt < 4; ++nt)
#pragma unroll
    for (int r = 0; r < 8; ++r)
      hbuf[rb + r][nt * 16 + cb] = (_Float16)siluf(acc2[nt][r]);
}

__global__ __launch_bounds__(WPB * 32) void final_kernel(
    const float* __restrict__ v_in, const int* __restrict__ attr,
    const float* __restrict__ scale, const float* __restrict__ shift,
    const _Float16* __restrict__ pwf1, const _Float16* __restrict__ pwf2,
    const float* __restrict__ Wf3,
    const _Float16* __restrict__ pwfg1, const _Float16* __restrict__ pwfg2,
    const float* __restrict__ Wfg3,
    float* __restrict__ out, int N, int APG) {
  __shared__ _Float16 h_s[WPB][16][64];
  __shared__ float core_s[WPB][16];
  int wave = threadIdx.x >> 5, lane = threadIdx.x & 31;
  int a0 = (blockIdx.x * WPB + wave) * 16;
  if (a0 >= N) return;

  mlp2_wave(v_in, a0, pwf1, pwf2, h_s[wave], lane);
  if (lane < 16) {
    float s = 0.0f;
#pragma unroll
    for (int c = 0; c < H; ++c) s += (float)h_s[wave][lane][c] * Wf3[c];
    core_s[wave][lane] = s;
  }
  mlp2_wave(v_in, a0, pwfg1, pwfg2, h_s[wave], lane);
  if (lane < 16) {
    float gdot = 0.0f;
#pragma unroll
    for (int c = 0; c < H; ++c) gdot += (float)h_s[wave][lane][c] * Wfg3[c];
    int a = a0 + lane;
    int z = attr[a];
    float ei = core_s[wave][lane] * sigm(gdot);
    ei = ei * scale[z] + shift[z];
    atomicAdd(&out[a / APG], ei);
  }
}

// ---------------------------------------------------------------------------
extern "C" void kernel_launch(void* const* d_in, const int* in_sizes, int n_in,
                              void* d_out, int out_size, void* d_ws, size_t ws_size,
                              hipStream_t stream) {
  const float* atom_pos   = (const float*)d_in[0];
  const float* cell       = (const float*)d_in[1];
  const float* pbc        = (const float*)d_in[2];
  const int*   atom_attr  = (const int*)d_in[3];
  const int*   edge_index = (const int*)d_in[4];
  const int*   tbi        = (const int*)d_in[5];
  const float* W_embed    = (const float*)d_in[11];
  const float* W_edge_enc = (const float*)d_in[12];
  const float* W3_atom    = (const float*)d_in[13];
  const float* W3_sbf     = (const float*)d_in[14];
  const float* W3_edge    = (const float*)d_in[15];
  const float* Wb1        = (const float*)d_in[16];
  const float* Wb2        = (const float*)d_in[17];
  const float* Wbg1       = (const float*)d_in[18];
  const float* Wbg2       = (const float*)d_in[19];
  const float* We0b       = (const float*)d_in[20];
  const float* Wa1        = (const float*)d_in[21];
  const float* Wa2        = (const float*)d_in[22];
  const float* Wag1       = (const float*)d_in[23];
  const float* Wag2       = (const float*)d_in[24];
  const float* We0a       = (const float*)d_in[25];
  const float* Wf1        = (const float*)d_in[26];
  const float* Wf2        = (const float*)d_in[27];
  const float* Wf3        = (const float*)d_in[28];
  const float* Wfg1       = (const float*)d_in[29];
  const float* Wfg2       = (const float*)d_in[30];
  const float* Wfg3       = (const float*)d_in[31];
  const float* scale      = (const float*)d_in[32];
  const float* shift      = (const float*)d_in[33];

  int N = in_sizes[0] / 3;
  int E = in_sizes[2] / 3;
  int T = in_sizes[5] / 2;
  int G = in_sizes[1] / 9;
  int APG = N / G, BPG = E / G, TPG = T / G;

  // workspace carve (256B aligned)
  size_t off = 0;
  char* base = (char*)d_ws;
  auto carve = [&](size_t bytes) -> void* {
    off = (off + 255) & ~(size_t)255;
    void* p = base + off;
    off += bytes;
    return p;
  };
  float* evec  = (float*)carve((size_t)E * 3 * 4);
  float* elen  = (float*)carve((size_t)E * 4);
  float* e0b   = (float*)carve((size_t)E * 4 * 4);
  float* fcr   = (float*)carve((size_t)E * 4);
  float* efeat = (float*)carve((size_t)E * H * 4);
  float* agg   = (float*)carve((size_t)E * H * 4);
  float* vA    = (float*)carve((size_t)N * H * 4);
  float* vB    = (float*)carve((size_t)N * H * 4);
  float* sbf   = (float*)carve((size_t)T * 16 * 4);
  float* fc3   = (float*)carve((size_t)T * 4);
  int*   t_ij  = (int*)carve((size_t)T * 4);
  int*   t_ak  = (int*)carve((size_t)T * 4);
  _Float16* pk = (_Float16*)carve((size_t)512 * 1024 * 2);

  // pre-pack all weights into WMMA-B fragment order (f16)
  _Float16* cur = pk;
  auto pack = [&](const float* W, int K, int Kpad) -> _Float16* {
    int frags = (Kpad / 32) * 4;
    _Float16* dst = cur;
    cur += (size_t)frags * 32 * 16;
    int total = frags * 32;
    prepack_kernel<<<(total + 255) / 256, 256, 0, stream>>>(W, dst, K, Kpad);
    return dst;
  };
  _Float16 *pw3atom[NLAYERS], *pw3sbf[NLAYERS], *pw3edge[NLAYERS],
      *pwb1[NLAYERS], *pwb2[NLAYERS], *pwbg1[NLAYERS], *pwbg2[NLAYERS],
      *pwa1[NLAYERS], *pwa2[NLAYERS], *pwag1[NLAYERS], *pwag2[NLAYERS];
  for (int l = 0; l < NLAYERS; ++l) {
    pw3atom[l] = pack(W3_atom + (size_t)l * H * H, 64, 64);
    pw3sbf[l]  = pack(W3_sbf + (size_t)l * 16 * H, 16, 32);
    pw3edge[l] = pack(W3_edge + (size_t)l * H * H, 64, 64);
    pwb1[l]  = pack(Wb1 + (size_t)l * 192 * H, 192, 192);
    pwb2[l]  = pack(Wb2 + (size_t)l * H * H, 64, 64);
    pwbg1[l] = pack(Wbg1 + (size_t)l * 192 * H, 192, 192);
    pwbg2[l] = pack(Wbg2 + (size_t)l * H * H, 64, 64);
    pwa1[l]  = pack(Wa1 + (size_t)l * 192 * H, 192, 192);
    pwa2[l]  = pack(Wa2 + (size_t)l * H * H, 64, 64);
    pwag1[l] = pack(Wag1 + (size_t)l * 192 * H, 192, 192);
    pwag2[l] = pack(Wag2 + (size_t)l * H * H, 64, 64);
  }
  _Float16* pwf1  = pack(Wf1, 64, 64);
  _Float16* pwf2  = pack(Wf2, 64, 64);
  _Float16* pwfg1 = pack(Wfg1, 64, 64);
  _Float16* pwfg2 = pack(Wfg2, 64, 64);

  // precompute
  edge_geom_kernel<<<(E + 255) / 256, 256, 0, stream>>>(
      atom_pos, cell, pbc, edge_index, edge_index + E, evec, elen, e0b, fcr, E, APG);
  edge_enc_kernel<<<(E * H + 255) / 256, 256, 0, stream>>>(e0b, W_edge_enc, efeat, E * H);
  atom_init_kernel<<<(N * H + 255) / 256, 256, 0, stream>>>(W_embed, atom_attr, vA, N * H);
  triple_init_kernel<<<(T + 255) / 256, 256, 0, stream>>>(
      tbi, evec, elen, fcr, edge_index + E, t_ij, t_ak, sbf, fc3, T, TPG, BPG);

  hipMemsetAsync(d_out, 0, (size_t)G * 4, stream);

  float* vcur = vA;
  float* vnxt = vB;
  for (int l = 0; l < NLAYERS; ++l) {
    hipMemsetAsync(agg, 0, (size_t)E * H * 4, stream);
    hipMemcpyAsync(vnxt, vcur, (size_t)N * H * 4, hipMemcpyDeviceToDevice, stream);
    int ttiles = T / 16;
    triple_kernel<<<(ttiles + WPB - 1) / WPB, WPB * 32, 0, stream>>>(
        vcur, sbf, fc3, t_ij, t_ak, agg, pw3atom[l], pw3sbf[l], T);
    EdgeW ew;
    ew.w3edge = pw3edge[l];
    ew.wb1 = pwb1[l];  ew.wb2 = pwb2[l];
    ew.wbg1 = pwbg1[l]; ew.wbg2 = pwbg2[l];
    ew.wa1 = pwa1[l];  ew.wa2 = pwa2[l];
    ew.wag1 = pwag1[l]; ew.wag2 = pwag2[l];
    ew.we0b = We0b + (size_t)l * 4 * H;
    ew.we0a = We0a + (size_t)l * 4 * H;
    int etiles = E / 16;
    edge_layer_kernel<<<(etiles + WPB - 1) / WPB, WPB * 32, 0, stream>>>(
        edge_index, edge_index + E, agg, efeat, vcur, vnxt, e0b, ew, E);
    float* tmp = vcur; vcur = vnxt; vnxt = tmp;
  }

  int atiles = N / 16;
  final_kernel<<<(atiles + WPB - 1) / WPB, WPB * 32, 0, stream>>>(
      vcur, atom_attr, scale, shift, pwf1, pwf2, Wf3, pwfg1, pwfg2, Wfg3,
      (float*)d_out, N, APG);
}
